// GraphAttentionLayer_36532991820033
// MI455X (gfx1250) — matrix operand order repaired and checked by reading, not compiled, not run on hardware
//
#include <hip/hip_runtime.h>
#include <hip/hip_bf16.h>

// CDNA5 / gfx1250 WMMA vector operand types
typedef float  v2f   __attribute__((ext_vector_type(2)));
typedef float  v8f   __attribute__((ext_vector_type(8)));
typedef __bf16 v16bf __attribute__((ext_vector_type(16)));

#define NEG_INF_F (-9000000000000000.0f)
#define LEAKY 0.01f

// Flat addresses of __shared__ objects carry the LDS byte offset in their low
// 32 bits (local->flat addrspacecast = {aperture, lds_offset}); async-to-LDS
// instructions take exactly that offset in the VDST VGPR.
__device__ __forceinline__ unsigned lds_addr_of(const void* p) {
  return (unsigned)(unsigned long long)p;
}

__device__ __forceinline__ void async_load_b32(unsigned lds, const void* gptr) {
  asm volatile("global_load_async_to_lds_b32 %0, %1, off"
               :: "v"(lds), "v"((unsigned long long)gptr) : "memory");
}
__device__ __forceinline__ void async_load_b64(unsigned lds, const void* gptr) {
  asm volatile("global_load_async_to_lds_b64 %0, %1, off"
               :: "v"(lds), "v"((unsigned long long)gptr) : "memory");
}
__device__ __forceinline__ void async_load_b128(unsigned lds, const void* gptr) {
  asm volatile("global_load_async_to_lds_b128 %0, %1, off"
               :: "v"(lds), "v"((unsigned long long)gptr) : "memory");
}
__device__ __forceinline__ void wait_async0() {
  asm volatile("s_wait_asynccnt 0x0" ::: "memory");
}

// ---------------------------------------------------------------------------
// Kernel 0: zero the output (split-K GEMM accumulates with float atomics).
// ---------------------------------------------------------------------------
__global__ void gat_zero(float* __restrict__ out, int n4) {
  int i = blockIdx.x * blockDim.x + threadIdx.x;
  if (i < n4) ((float4*)out)[i] = make_float4(0.f, 0.f, 0.f, 0.f);
}

// ---------------------------------------------------------------------------
// Kernel 1: h = X @ W in exact fp32 via V_WMMA_F32_16X16X4_F32.
//   grid = N/128 blocks x 256 threads (8 waves, one 16-row M-tile per wave).
//   W staged in LDS in two 64-row halves via GLOBAL_LOAD_ASYNC_TO_LDS_B128
//   (ASYNCcnt-tracked, no VGPR round trip). Row stride 136 dwords: B-operand
//   reads of lanes 0-15 (row k) vs 16-31 (row k+2) are 2*136 mod 64 = 16
//   banks apart -> conflict-free halves; rows stay 16B aligned for b128.
//   Writes h both as f32 (exact logits) and bf16 (GEMM B operand).
// ---------------------------------------------------------------------------
__global__ void gat_h_kernel(const float* __restrict__ X,
                             const float* __restrict__ W,
                             float* __restrict__ hf,
                             __bf16* __restrict__ hb,
                             int N) {
  constexpr int F = 128;
  constexpr int WS = 136;                 // LDS row stride (dwords)
  __shared__ __align__(16) float Ws[64 * WS];

  const int  lane = threadIdx.x & 31;
  const int  wave = threadIdx.x >> 5;
  const bool hi   = lane >= 16;
  const int  row  = lane & 15;            // A-matrix M / B-matrix N index
  const int  i0   = blockIdx.x * 128 + wave * 16;

  const float* xrow = X + (size_t)(i0 + row) * F;

  v8f acc[8] = {};                        // 16x128 f32 C tile (8 N-tiles)

  for (int kh = 0; kh < F; kh += 64) {
    __syncthreads();  // previous half fully consumed
    // async-stage W[kh:kh+64][0:128] -> LDS, 16B per lane per instruction
    for (int idx = threadIdx.x; idx < 64 * 32; idx += 256) {
      const int r = idx >> 5, c4 = idx & 31;   // c4 in float4 units
      async_load_b128(lds_addr_of(&Ws[r * WS + c4 * 4]),
                      W + (size_t)(kh + r) * F + c4 * 4);
    }
    wait_async0();
    __syncthreads();

    for (int k = 0; k < 64; k += 4) {
      // A 16x4 f32: vgpr0 = K (k | k+2 by lane half), vgpr1 = K+1
      const int kA = k + (hi ? 2 : 0);
      v2f A;
      A.x = xrow[kh + kA];
      A.y = xrow[kh + kA + 1];
#pragma unroll
      for (int t = 0; t < 8; t++) {
        v2f B;
        B.x = Ws[kA * WS + t * 16 + row];
        B.y = Ws[(kA + 1) * WS + t * 16 + row];
        acc[t] = __builtin_amdgcn_wmma_f32_16x16x4_f32(
            false, A, false, B, (short)0, acc[t], false, false);
      }
    }
  }

  // C/D layout: vgpr v holds M=v (lanes0-15) and M=v+8 (lanes16-31), N=lane&15
#pragma unroll
  for (int t = 0; t < 8; t++)
#pragma unroll
    for (int v = 0; v < 8; v++) {
      const int   orow = i0 + v + (hi ? 8 : 0);
      const int   ocol = t * 16 + row;
      const float val  = acc[t][v];
      hf[(size_t)orow * F + ocol] = val;
      hb[(size_t)orow * F + ocol] = (__bf16)val;
    }
}

// ---------------------------------------------------------------------------
// Kernel 2: s1 = h@a1, s2 = h@a2 (one row per wave, shuffle reduction).
// ---------------------------------------------------------------------------
__global__ void gat_s12_kernel(const float* __restrict__ hf,
                               const float* __restrict__ a,
                               float* __restrict__ s1,
                               float* __restrict__ s2,
                               int N) {
  constexpr int F = 128;
  const int lane = threadIdx.x & 31;
  const int wave = threadIdx.x >> 5;
  const int row  = blockIdx.x * (blockDim.x >> 5) + wave;
  if (row >= N) return;

  float p1 = 0.f, p2 = 0.f;
  const float* hr = hf + (size_t)row * F;
  for (int c = lane; c < F; c += 32) {
    const float v = hr[c];
    p1 += v * a[c];
    p2 += v * a[F + c];
  }
  for (int off = 16; off >= 1; off >>= 1) {
    p1 += __shfl_xor(p1, off);
    p2 += __shfl_xor(p2, off);
  }
  if (lane == 0) { s1[row] = p1; s2[row] = p2; }
}

// ---------------------------------------------------------------------------
// Kernel 3: per-row online softmax statistics (max m_i, sum l_i).
//   One block per row; single coalesced read of the adj row. The reference
//   masks with a *finite* -9e15, so every column participates; arithmetic
//   stays finite (start m at -FLT_MAX, exp underflows cleanly to 0).
// ---------------------------------------------------------------------------
__global__ void gat_rowstats_kernel(const int* __restrict__ adj,
                                    const float* __restrict__ s1,
                                    const float* __restrict__ s2,
                                    float* __restrict__ mrow,
                                    float* __restrict__ lrow,
                                    int N) {
  const int    i    = blockIdx.x;
  const float  s1v  = s1[i];
  const int*   arow = adj + (size_t)i * N;

  float m = -3.0e38f, l = 0.f;
  for (int j = threadIdx.x; j < N; j += blockDim.x) {
    __builtin_prefetch(arow + j + 4 * blockDim.x, 0, 1);  // global_prefetch_b8
    const int   ad = arow[j];
    const float x  = s1v + s2[j];
    const float e  = (ad > 0) ? (x > 0.f ? x : LEAKY * x) : NEG_INF_F;
    const float nm = fmaxf(m, e);
    l = l * __expf(m - nm) + __expf(e - nm);
    m = nm;
  }

  const int lane = threadIdx.x & 31;
  const int wave = threadIdx.x >> 5;
  for (int off = 16; off >= 1; off >>= 1) {
    const float om = __shfl_xor(m, off);
    const float ol = __shfl_xor(l, off);
    const float nm = fmaxf(m, om);
    l = l * __expf(m - nm) + ol * __expf(om - nm);
    m = nm;
  }
  __shared__ float sm[8], sl[8];
  if (lane == 0) { sm[wave] = m; sl[wave] = l; }
  __syncthreads();
  if (threadIdx.x == 0) {
    float M = sm[0], L = sl[0];
    const int nw = blockDim.x >> 5;
    for (int w = 1; w < nw; w++) {
      const float nm = fmaxf(M, sm[w]);
      L = L * __expf(M - nm) + sl[w] * __expf(sm[w] - nm);
      M = nm;
    }
    mrow[i] = M;
    lrow[i] = L;
  }
}

// ---------------------------------------------------------------------------
// Kernel 4: out += P @ h via V_WMMA_F32_16X16X32_BF16 with double-buffered
//   async-to-LDS staging. p_ij = exp(e_ij - m_i)/l_i computed on the fly;
//   m/l precomputed so split-K partials sum exactly.
//   Block = 128 threads (4 waves x 16 rows). j chunked by 64.
//   hB stride 68 bf16 (136B: 8B-aligned rows for b64 async; 8*68 bf16 =
//   272 dw mod 64 = 16 -> lane halves of B reads hit disjoint bank groups).
//   Pipeline per chunk: s_wait_asynccnt 0 -> barrier -> issue next chunk's
//   async loads into the other buffer -> compute current buffer.
// ---------------------------------------------------------------------------
__global__ void gat_attn_kernel(const int* __restrict__ adj,
                                const __bf16* __restrict__ hb,
                                const float* __restrict__ s1,
                                const float* __restrict__ s2,
                                const float* __restrict__ mrow,
                                const float* __restrict__ lrow,
                                float* __restrict__ out,
                                int N, int jSpan) {
  constexpr int F  = 128;
  constexpr int HS = 68;                  // hB row stride in bf16
  __shared__ __align__(16) __bf16 hB[2][64 * HS];
  __shared__ __align__(16) int    adjT[2][4][16][68];
  __shared__ __align__(16) float  s2s[2][64];

  const int  lane = threadIdx.x & 31;
  const int  wave = threadIdx.x >> 5;
  const bool hi   = lane >= 16;
  const int  row  = lane & 15;            // A-matrix M / B-matrix N index
  const int  i0w  = blockIdx.x * 64 + wave * 16;
  const int  gi   = i0w + row;

  const float s1v  = s1[gi];
  const float mv   = mrow[gi];
  const float invl = 1.0f / lrow[gi];     // l >= 1 always (finite mask value)

  const int jbeg = blockIdx.y * jSpan;
  const int jend = jbeg + jSpan;

  // Issue all async loads staging chunk j0 into buffer b.
  auto stage = [&](int j0, int b) {
    // h[j0:j0+64][0:128] bf16 -> hB[b]; 8B per lane per instruction
    for (int idx = threadIdx.x; idx < 64 * 32; idx += 128) {
      const int r = idx >> 5, c = idx & 31;      // c in 4-bf16 (8B) units
      async_load_b64(lds_addr_of(&hB[b][r * HS + c * 4]),
                     hb + (size_t)(j0 + r) * F + c * 4);
    }
    // this wave's adj tile [16 rows][64 cols] -> adjT[b][wave]; 16B per lane
    for (int idx = lane; idx < 16 * 16; idx += 32) {
      const int r = idx >> 4, c4 = idx & 15;
      async_load_b128(lds_addr_of(&adjT[b][wave][r][c4 * 4]),
                      adj + (size_t)(i0w + r) * N + j0 + c4 * 4);
    }
    if (threadIdx.x < 64)
      async_load_b32(lds_addr_of(&s2s[b][threadIdx.x]), s2 + j0 + threadIdx.x);
  };

  v8f acc[8] = {};
  stage(jbeg, 0);
  int buf = 0;

  for (int j0 = jbeg; j0 < jend; j0 += 64, buf ^= 1) {
    wait_async0();        // own async loads into `buf` landed
    __syncthreads();      // everyone's landed; previous compute finished
    if (j0 + 64 < jend) stage(j0 + 64, buf ^ 1);

#pragma unroll
    for (int kk = 0; kk < 64; kk += 32) {
      // A (16x32 bf16): lane holds row=lane&15; element e -> K per ISA layout:
      //   K = (e&7) + (e>=8 ? 16 : 0) + (lane>=16 ? 8 : 0)
      v16bf A;
#pragma unroll
      for (int e = 0; e < 16; e++) {
        const int   K  = (e & 7) + ((e & 8) ? 16 : 0) + (hi ? 8 : 0);
        const int   ad = adjT[buf][wave][row][kk + K];
        const float x  = s1v + s2s[buf][kk + K];
        const float ev = (ad > 0) ? (x > 0.f ? x : LEAKY * x) : NEG_INF_F;
        const float p  = __expf(ev - mv) * invl;
        A[e] = (__bf16)p;
      }
#pragma unroll
      for (int t = 0; t < 8; t++) {
        // B (32x16 bf16): col=lane&15; element e -> K = (e&7) + (e>=8 ? 16 : 0) + (lane>=16 ? 8 : 0)
        v16bf B;
#pragma unroll
        for (int e = 0; e < 16; e++) {
          const int K = (e & 7) + ((e & 8) ? 16 : 0) + (hi ? 8 : 0);
          B[e] = hB[buf][(kk + K) * HS + t * 16 + row];
        }
        acc[t] = __builtin_amdgcn_wmma_f32_16x16x32_bf16(
            false, A, false, B, (short)0, acc[t], false, false);
      }
    }
  }

  // accumulate split-K partials
#pragma unroll
  for (int t = 0; t < 8; t++)
#pragma unroll
    for (int v = 0; v < 8; v++) {
      const int orow = i0w + v + (hi ? 8 : 0);
      atomicAdd(&out[(size_t)orow * F + t * 16 + row], acc[t][v]);
    }
}

// ---------------------------------------------------------------------------
// Host launcher
// ---------------------------------------------------------------------------
extern "C" void kernel_launch(void* const* d_in, const int* in_sizes, int n_in,
                              void* d_out, int out_size, void* d_ws, size_t ws_size,
                              hipStream_t stream) {
  const float* X   = (const float*)d_in[0];
  const int*   adj = (const int*)d_in[1];
  const float* W   = (const float*)d_in[2];
  const float* a   = (const float*)d_in[3];
  float*       out = (float*)d_out;

  const int F = in_sizes[3] / 2;   // a has shape (2F,1) -> F = 128
  const int N = in_sizes[0] / F;   // 8192

  // workspace layout: hf (N*F f32) | s1 | s2 | m | l | hb (N*F bf16)
  float*  hf   = (float*)d_ws;
  float*  s1   = hf + (size_t)N * F;
  float*  s2   = s1 + N;
  float*  mrow = s2 + N;
  float*  lrow = mrow + N;
  __bf16* hb   = (__bf16*)(lrow + N);

  gat_zero<<<dim3((N * F / 4 + 255) / 256), dim3(256), 0, stream>>>(out, N * F / 4);
  gat_h_kernel<<<dim3(N / 128), dim3(256), 0, stream>>>(X, W, hf, hb, N);
  gat_s12_kernel<<<dim3(N / 8), dim3(256), 0, stream>>>(hf, a, s1, s2, N);
  gat_rowstats_kernel<<<dim3(N), dim3(256), 0, stream>>>(adj, s1, s2, mrow, lrow, N);

  const int SPLITS = 8;  // split-K over j for occupancy; partials sum exactly
  gat_attn_kernel<<<dim3(N / 64, SPLITS), dim3(128), 0, stream>>>(
      adj, hb, s1, s2, mrow, lrow, out, N, N / SPLITS);
}
